// DecisionTransformer_34230889349510
// MI455X (gfx1250) — matrix operand order repaired and checked by reading, not compiled, not run on hardware
//
#include <hip/hip_runtime.h>
#include <hip/hip_bf16.h>

typedef _Float16 v16h __attribute__((ext_vector_type(16)));
typedef _Float16 v8h  __attribute__((ext_vector_type(8)));
typedef float    v8f  __attribute__((ext_vector_type(8)));

#define kB   8
#define kT   64
#define kTok 18
#define kS   1152      // kT * kTok
#define kH   768
#define kFF  3072
#define kNH  12
#define kHD  64
#define kM   9216      // kB * kS
#define NEGV (-10000.0f)

// ---------------- WMMA fragment loaders (CDNA5 documented layouts) ----------
// A (16x32 f16): lanes 0-15 row M=lane, halves j<8 -> K=j, j>=8 -> K=16+j-8
//                lanes 16-31 row M=lane-16, K base 8 / 24.
__device__ __forceinline__ v16h load_frag_a(const _Float16* base, int ld,
                                            int row0, int k0, int lane) {
    int m  = row0 + (lane & 15);
    int kb = (lane < 16) ? 0 : 8;
    const _Float16* p = base + (size_t)m * ld + k0 + kb;
    v8h lo = *(const v8h*)p;
    v8h hi = *(const v8h*)(p + 16);
    return __builtin_shufflevector(lo, hi, 0,1,2,3,4,5,6,7,8,9,10,11,12,13,14,15);
}
// B (32x16 f16), B stored as N x K row-major: lanes 0-15 col N=lane hold K 0-7,16-23,
// lanes 16-31 col N=lane-16 hold K 8-15,24-31 (same order as A).
__device__ __forceinline__ v16h load_frag_b(const _Float16* base, int ld,
                                            int n0, int k0, int lane) {
    int n  = n0 + (lane & 15);
    int kb = (lane < 16) ? 0 : 8;
    const _Float16* p = base + (size_t)n * ld + k0 + kb;
    v8h lo = *(const v8h*)p;
    v8h hi = *(const v8h*)(p + 16);
    return __builtin_shufflevector(lo, hi, 0,1,2,3,4,5,6,7,8,9,10,11,12,13,14,15);
}
__device__ __forceinline__ v8f wmma_f16(v16h a, v16h b, v8f c) {
    return __builtin_amdgcn_wmma_f32_16x16x32_f16(false, a, false, b,
                                                  (short)0, c, false, false);
}

// ---------------- weight fp32 (K,N) -> f16 (N,K) -----------------------------
__global__ void transpose_to_f16(const float* __restrict__ src,
                                 _Float16* __restrict__ dst, int Kd, int Nd) {
    size_t e   = (size_t)blockIdx.x * blockDim.x + threadIdx.x;
    size_t tot = (size_t)Kd * Nd;
    if (e >= tot) return;
    int i = (int)(e / Nd), o = (int)(e % Nd);
    dst[(size_t)o * Kd + i] = (_Float16)src[e];
}

// ---------------- embedding + embed LN + key-mask ----------------------------
__global__ __launch_bounds__(128)
void embed_kernel(const float* __restrict__ images, const int* __restrict__ missions,
                  const float* __restrict__ mmask, const float* __restrict__ actions,
                  const float* __restrict__ rtg, const int* __restrict__ tsteps,
                  const int* __restrict__ amask,
                  const float* __restrict__ W_time, const float* __restrict__ W_pos,
                  const float* __restrict__ W_ret, const float* __restrict__ b_ret,
                  const float* __restrict__ W_img, const float* __restrict__ b_img,
                  const float* __restrict__ W_mis, const float* __restrict__ W_act,
                  const float* __restrict__ b_act,
                  const float* __restrict__ ln_g, const float* __restrict__ ln_b,
                  float* __restrict__ xout, float* __restrict__ mask_add) {
    __shared__ float red[128];
    int srow = blockIdx.x;               // 0..9215
    int b = srow / kS, s = srow % kS;
    int t = s / kTok, tok = s % kTok;
    int tid = threadIdx.x;
    int bt = b * kT + t;
    const float* te = W_time + (size_t)tsteps[bt] * kH;

    float vals[6];
#pragma unroll
    for (int i = 0; i < 6; ++i) {
        int hc = tid + i * 128;
        float v = te[hc];
        if (tok == 0) {
            v += rtg[bt] * W_ret[hc] + b_ret[hc];
        } else if (tok <= 3) {
            int c = tok - 1;
            float a2 = b_img[hc];
            const float* ib = images + (size_t)bt * 49 * 3 + c;
            for (int p = 0; p < 49; ++p) a2 += ib[p * 3] * W_img[p * kH + hc];
            v += a2 + W_pos[(tok - 1) * kH + hc];
        } else if (tok <= 16) {
            int j = tok - 4;
            int mt = missions[bt * 13 + j];
            v += W_mis[mt * kH + hc] + W_pos[(tok - 1) * kH + hc];
        } else {
            float a2 = b_act[hc];
            const float* ab = actions + (size_t)bt * 7;
            for (int a = 0; a < 7; ++a) a2 += ab[a] * W_act[a * kH + hc];
            v += a2;
        }
        vals[i] = v;
    }
    float ps = 0.f;
#pragma unroll
    for (int i = 0; i < 6; ++i) ps += vals[i];
    red[tid] = ps; __syncthreads();
    for (int off = 64; off > 0; off >>= 1) {
        if (tid < off) red[tid] += red[tid + off];
        __syncthreads();
    }
    float mean = red[0] * (1.0f / kH);
    __syncthreads();
    float pv = 0.f;
#pragma unroll
    for (int i = 0; i < 6; ++i) { float d = vals[i] - mean; pv += d * d; }
    red[tid] = pv; __syncthreads();
    for (int off = 64; off > 0; off >>= 1) {
        if (tid < off) red[tid] += red[tid + off];
        __syncthreads();
    }
    float rstd = rsqrtf(red[0] * (1.0f / kH) + 1e-5f);
    float* xr = xout + (size_t)srow * kH;
#pragma unroll
    for (int i = 0; i < 6; ++i) {
        int hc = tid + i * 128;
        xr[hc] = (vals[i] - mean) * rstd * ln_g[hc] + ln_b[hc];
    }
    if (tid == 0) {
        float tm = 1.f;
        if (tok >= 4 && tok <= 16) tm = mmask[bt * 13 + (tok - 4)];
        mask_add[srow] = (1.f - (float)amask[bt] * tm) * NEGV;
    }
}

// ---------------- row LayerNorm fp32 -> f16 ----------------------------------
__global__ __launch_bounds__(256)
void ln_kernel(const float* __restrict__ x, const float* __restrict__ g,
               const float* __restrict__ be, _Float16* __restrict__ out) {
    int wave = threadIdx.x >> 5, lane = threadIdx.x & 31;
    int row = blockIdx.x * 8 + wave;
    const float* xr = x + (size_t)row * kH;
    float vals[24]; float s = 0.f;
#pragma unroll
    for (int i = 0; i < 24; ++i) { vals[i] = xr[lane + i * 32]; s += vals[i]; }
    for (int o = 16; o > 0; o >>= 1) s += __shfl_xor(s, o, 32);
    float mean = s * (1.0f / kH);
    float v = 0.f;
#pragma unroll
    for (int i = 0; i < 24; ++i) { float d = vals[i] - mean; v += d * d; }
    for (int o = 16; o > 0; o >>= 1) v += __shfl_xor(v, o, 32);
    float rstd = rsqrtf(v * (1.0f / kH) + 1e-5f);
    _Float16* orow = out + (size_t)row * kH;
#pragma unroll
    for (int i = 0; i < 24; ++i) {
        int c = lane + i * 32;
        orow[c] = (_Float16)((vals[i] - mean) * rstd * g[c] + be[c]);
    }
}

// ---------------- generic WMMA GEMM: C = A(MxK) * Bt(NxK)^T + bias -----------
// Software-pipelined K-loop: next iteration's fragments are loaded into fresh
// registers before the current WMMAs issue, so global_load_b128 clauses overlap
// with matrix-op execution instead of draining loadcnt to 0 every step.
// mode 0: store f16 row-major   1: GELU -> f16   2: += into fp32 residual
// mode 3: QKV scatter (B,NH,S,HD)   4: V scatter transposed (B,NH,HD,S)
__global__ __launch_bounds__(256)
void gemm_wmma(const _Float16* __restrict__ A, const _Float16* __restrict__ Bt,
               const float* __restrict__ bias, _Float16* __restrict__ outH,
               float* __restrict__ resid, int N, int K, int mode) {
    int wave = threadIdx.x >> 5, lane = threadIdx.x & 31;
    int row0 = (blockIdx.x * 8 + wave) * 16;
    int col0 = blockIdx.y * 64;

    // warm L0/L2 for this wave's weight panel (global_prefetch_b8)
    __builtin_prefetch(Bt + (size_t)(col0 + (lane & 15)) * K, 0, 1);
    __builtin_prefetch(Bt + (size_t)(col0 + 16 + (lane & 15)) * K, 0, 1);
    __builtin_prefetch(Bt + (size_t)(col0 + 32 + (lane & 15)) * K, 0, 1);
    __builtin_prefetch(Bt + (size_t)(col0 + 48 + (lane & 15)) * K, 0, 1);

    v8f acc[4] = {};
    v16h a_c  = load_frag_a(A, K, row0, 0, lane);
    v16h b_c0 = load_frag_b(Bt, K, col0 +  0, 0, lane);
    v16h b_c1 = load_frag_b(Bt, K, col0 + 16, 0, lane);
    v16h b_c2 = load_frag_b(Bt, K, col0 + 32, 0, lane);
    v16h b_c3 = load_frag_b(Bt, K, col0 + 48, 0, lane);
    for (int k0 = 32; k0 < K; k0 += 32) {
        v16h a_n  = load_frag_a(A, K, row0, k0, lane);
        v16h b_n0 = load_frag_b(Bt, K, col0 +  0, k0, lane);
        v16h b_n1 = load_frag_b(Bt, K, col0 + 16, k0, lane);
        v16h b_n2 = load_frag_b(Bt, K, col0 + 32, k0, lane);
        v16h b_n3 = load_frag_b(Bt, K, col0 + 48, k0, lane);
        acc[0] = wmma_f16(a_c, b_c0, acc[0]);
        acc[1] = wmma_f16(a_c, b_c1, acc[1]);
        acc[2] = wmma_f16(a_c, b_c2, acc[2]);
        acc[3] = wmma_f16(a_c, b_c3, acc[3]);
        a_c  = a_n;
        b_c0 = b_n0; b_c1 = b_n1; b_c2 = b_n2; b_c3 = b_n3;
    }
    acc[0] = wmma_f16(a_c, b_c0, acc[0]);
    acc[1] = wmma_f16(a_c, b_c1, acc[1]);
    acc[2] = wmma_f16(a_c, b_c2, acc[2]);
    acc[3] = wmma_f16(a_c, b_c3, acc[3]);

    int nloc = lane & 15, mo = (lane < 16) ? 0 : 8;
#pragma unroll
    for (int j = 0; j < 4; ++j) {
        int col = col0 + j * 16 + nloc;
        float bv = bias[col];
#pragma unroll
        for (int r = 0; r < 8; ++r) {
            int row = row0 + mo + r;
            float val = acc[j][r] + bv;
            if (mode == 0) {
                outH[(size_t)row * N + col] = (_Float16)val;
            } else if (mode == 1) {
                float gl = 0.5f * val * (1.f + erff(val * 0.70710678f));
                outH[(size_t)row * N + col] = (_Float16)gl;
            } else if (mode == 2) {
                resid[(size_t)row * N + col] += val;
            } else {
                int b = row / kS, s = row % kS;
                int head = col >> 6, d = col & 63;
                size_t o = (mode == 3)
                    ? ((((size_t)b * kNH + head) * kS + s) * kHD + d)
                    : ((((size_t)b * kNH + head) * kHD + d) * kS + s);
                outH[o] = (_Float16)val;
            }
        }
    }
}

// ---------------- fused attention: one wave per (b, head, 16-query tile) -----
__global__ __launch_bounds__(32)
void attn_kernel(const _Float16* __restrict__ q, const _Float16* __restrict__ k,
                 const _Float16* __restrict__ vT, const float* __restrict__ mask_add,
                 _Float16* __restrict__ y) {
    __shared__ __align__(16) _Float16 sc[16 * kS];   // 36 KB score/prob tile
    __shared__ float rowmax[16], rowsum[16];
    int lane = threadIdx.x;
    int bid = blockIdx.x;
    int qt = bid % (kS / 16); int tmp = bid / (kS / 16);
    int h = tmp % kNH, b = tmp / kNH;
    const _Float16* qb = q  + ((size_t)b * kNH + h) * kS * kHD;
    const _Float16* kb = k  + ((size_t)b * kNH + h) * kS * kHD;
    const _Float16* vb = vT + ((size_t)b * kNH + h) * kHD * kS;
    int nloc = lane & 15, mo = (lane < 16) ? 0 : 8;

    v16h q0 = load_frag_a(qb, kHD, qt * 16, 0,  lane);
    v16h q1 = load_frag_a(qb, kHD, qt * 16, 32, lane);

    // ---- scores: pipelined over key tiles; mask/convert/store overlaps loads
    v16h kc0 = load_frag_b(kb, kHD, 0, 0,  lane);
    v16h kc1 = load_frag_b(kb, kHD, 0, 32, lane);
    for (int kt = 0; kt < kS / 16; ++kt) {
        v16h kn0 = kc0, kn1 = kc1;
        if (kt + 1 < kS / 16) {
            kn0 = load_frag_b(kb, kHD, (kt + 1) * 16, 0,  lane);
            kn1 = load_frag_b(kb, kHD, (kt + 1) * 16, 32, lane);
        }
        v8f a = {};
        a = wmma_f16(q0, kc0, a);
        a = wmma_f16(q1, kc1, a);
        int kg = kt * 16 + nloc;
        float madd = mask_add[(size_t)b * kS + kg];
#pragma unroll
        for (int r = 0; r < 8; ++r) {
            int qg = qt * 16 + mo + r;
            float sv = (kg > qg) ? NEGV : a[r] * 0.125f;
            sc[(mo + r) * kS + kg] = (_Float16)(sv + madd);
        }
        kc0 = kn0; kc1 = kn1;
    }
    __syncthreads();
    {   // two lanes per query row: max then sum(exp)
        int row = lane >> 1, half = lane & 1;
        const _Float16* sr = &sc[row * kS + half * (kS / 2)];
        float mx = -3.0e38f;
        for (int i = 0; i < kS / 2; ++i) mx = fmaxf(mx, (float)sr[i]);
        mx = fmaxf(mx, __shfl_xor(mx, 1, 32));
        float sum = 0.f;
        for (int i = 0; i < kS / 2; ++i) sum += __expf((float)sr[i] - mx);
        sum += __shfl_xor(sum, 1, 32);
        if (half == 0) { rowmax[row] = mx; rowsum[row] = sum; }
    }
    __syncthreads();
    for (int e = lane; e < 16 * kS; e += 32) {
        int row = e / kS;
        sc[e] = (_Float16)(__expf((float)sc[e] - rowmax[row]) / rowsum[row]);
    }
    __syncthreads();
    // ---- P(16 x S) @ V(S x 64): A from LDS, pipelined V-fragment loads
    v8f acc[4] = {};
    v16h vc0 = load_frag_b(vb, kS,  0, 0, lane);
    v16h vc1 = load_frag_b(vb, kS, 16, 0, lane);
    v16h vc2 = load_frag_b(vb, kS, 32, 0, lane);
    v16h vc3 = load_frag_b(vb, kS, 48, 0, lane);
    for (int kt = 0; kt < kS / 32; ++kt) {
        int m = lane & 15, kbase = (lane < 16) ? 0 : 8;
        const _Float16* pa = &sc[m * kS + kt * 32 + kbase];
        v8h lo = *(const v8h*)pa;
        v8h hi = *(const v8h*)(pa + 16);
        v16h af = __builtin_shufflevector(lo, hi, 0,1,2,3,4,5,6,7,8,9,10,11,12,13,14,15);
        v16h vn0 = vc0, vn1 = vc1, vn2 = vc2, vn3 = vc3;
        if (kt + 1 < kS / 32) {
            int kn = (kt + 1) * 32;
            vn0 = load_frag_b(vb, kS,  0, kn, lane);
            vn1 = load_frag_b(vb, kS, 16, kn, lane);
            vn2 = load_frag_b(vb, kS, 32, kn, lane);
            vn3 = load_frag_b(vb, kS, 48, kn, lane);
        }
        acc[0] = wmma_f16(af, vc0, acc[0]);
        acc[1] = wmma_f16(af, vc1, acc[1]);
        acc[2] = wmma_f16(af, vc2, acc[2]);
        acc[3] = wmma_f16(af, vc3, acc[3]);
        vc0 = vn0; vc1 = vn1; vc2 = vn2; vc3 = vn3;
    }
    size_t ybase = (size_t)b * kS * kH;
#pragma unroll
    for (int j = 0; j < 4; ++j) {
        int d = h * kHD + j * 16 + nloc;
#pragma unroll
        for (int r = 0; r < 8; ++r) {
            int srow = qt * 16 + mo + r;
            y[ybase + (size_t)srow * kH + d] = (_Float16)acc[j][r];
        }
    }
}

// ---------------- gather last-state token + action head ---------------------
__global__ __launch_bounds__(32)
void predict_kernel(const float* __restrict__ x, const float* __restrict__ mmask,
                    const float* __restrict__ W_pred, const float* __restrict__ b_pred,
                    float* __restrict__ out) {
    int bt = blockIdx.x;                 // 0..511
    int lane = threadIdx.x;
    float valid = 0.f;
    for (int j = 0; j < 13; ++j) valid += mmask[bt * 13 + j];
    int idx = (int)(valid + 2.0f);       // valid + IMG_CH - 1
    int srow = bt * kTok + 1 + idx;
    const float* xr = x + (size_t)srow * kH;
    float acc[7] = {0, 0, 0, 0, 0, 0, 0};
    for (int hc = lane; hc < kH; hc += 32) {
        float xv = xr[hc];
#pragma unroll
        for (int a = 0; a < 7; ++a) acc[a] += xv * W_pred[hc * 7 + a];
    }
#pragma unroll
    for (int a = 0; a < 7; ++a)
        for (int o = 16; o > 0; o >>= 1) acc[a] += __shfl_xor(acc[a], o, 32);
    if (lane < 7) out[(size_t)bt * 7 + lane] = acc[lane] + b_pred[lane];
}

// ---------------- host orchestration ----------------------------------------
extern "C" void kernel_launch(void* const* d_in, const int* in_sizes, int n_in,
                              void* d_out, int out_size, void* d_ws, size_t ws_size,
                              hipStream_t stream) {
    const float* images   = (const float*)d_in[0];
    const int*   missions = (const int*)d_in[1];
    const float* mmask    = (const float*)d_in[2];
    const float* actions  = (const float*)d_in[3];
    const float* rtg      = (const float*)d_in[4];
    const int*   tsteps   = (const int*)d_in[5];
    const int*   amask    = (const int*)d_in[6];
    const float* W_time = (const float*)d_in[7];
    const float* W_pos  = (const float*)d_in[8];
    const float* W_ret  = (const float*)d_in[9];
    const float* b_ret  = (const float*)d_in[10];
    const float* W_img  = (const float*)d_in[11];
    const float* b_img  = (const float*)d_in[12];
    const float* W_mis  = (const float*)d_in[13];
    const float* W_act  = (const float*)d_in[14];
    const float* b_act  = (const float*)d_in[15];
    const float* lneg   = (const float*)d_in[16];
    const float* lneb   = (const float*)d_in[17];
    const float* ln1_g  = (const float*)d_in[18];
    const float* ln1_b  = (const float*)d_in[19];
    const float* ln2_g  = (const float*)d_in[20];
    const float* ln2_b  = (const float*)d_in[21];
    const float* Wq = (const float*)d_in[22];
    const float* bq = (const float*)d_in[23];
    const float* Wk = (const float*)d_in[24];
    const float* bk = (const float*)d_in[25];
    const float* Wv = (const float*)d_in[26];
    const float* bv = (const float*)d_in[27];
    const float* Wp = (const float*)d_in[28];
    const float* bp = (const float*)d_in[29];
    const float* W1 = (const float*)d_in[30];
    const float* b1 = (const float*)d_in[31];
    const float* W2 = (const float*)d_in[32];
    const float* b2 = (const float*)d_in[33];
    const float* W_pred = (const float*)d_in[34];
    const float* b_pred = (const float*)d_in[35];

    // workspace layout (all sizes multiples of 256 B)
    char* base = (char*)d_ws;
    size_t off = 0;
    float*    X   = (float*)(base + off);    off += (size_t)kM * kH * 4;   // residual
    _Float16* HN  = (_Float16*)(base + off); off += (size_t)kM * kH * 2;   // LN out
    _Float16* Qb  = (_Float16*)(base + off); off += (size_t)kM * kH * 2;
    _Float16* Kb  = (_Float16*)(base + off); off += (size_t)kM * kH * 2;
    _Float16* VT  = (_Float16*)(base + off); off += (size_t)kM * kH * 2;
    _Float16* Y   = (_Float16*)(base + off); off += (size_t)kM * kH * 2;
    _Float16* A1  = (_Float16*)(base + off); off += (size_t)kM * kFF * 2;
    _Float16* WT  = (_Float16*)(base + off); off += (size_t)4 * 7077888 * 2;
    float*    MADD = (float*)(base + off);   off += (size_t)kM * 4;
    (void)in_sizes; (void)n_in; (void)out_size; (void)ws_size;

    const size_t HH = (size_t)kH * kH;         // 589824
    const size_t HF = (size_t)kH * kFF;        // 2359296

    // one-pass weight conversion fp32 -> f16 transposed (L2-resident afterwards)
    for (int l = 0; l < 4; ++l) {
        _Float16* wl = WT + (size_t)l * 7077888;
        transpose_to_f16<<<(int)((HH + 255) / 256), 256, 0, stream>>>(Wq + l * HH, wl + 0 * HH, kH, kH);
        transpose_to_f16<<<(int)((HH + 255) / 256), 256, 0, stream>>>(Wk + l * HH, wl + 1 * HH, kH, kH);
        transpose_to_f16<<<(int)((HH + 255) / 256), 256, 0, stream>>>(Wv + l * HH, wl + 2 * HH, kH, kH);
        transpose_to_f16<<<(int)((HH + 255) / 256), 256, 0, stream>>>(Wp + l * HH, wl + 3 * HH, kH, kH);
        transpose_to_f16<<<(int)((HF + 255) / 256), 256, 0, stream>>>(W1 + l * HF, wl + 4 * HH, kH, kFF);
        transpose_to_f16<<<(int)((HF + 255) / 256), 256, 0, stream>>>(W2 + l * HF, wl + 4 * HH + HF, kFF, kH);
    }

    embed_kernel<<<kM, 128, 0, stream>>>(images, missions, mmask, actions, rtg,
                                         tsteps, amask, W_time, W_pos, W_ret, b_ret,
                                         W_img, b_img, W_mis, W_act, b_act,
                                         lneg, lneb, X, MADD);

    dim3 gH(kM / 128, kH / 64);    // N = 768
    dim3 gF(kM / 128, kFF / 64);   // N = 3072
    for (int l = 0; l < 4; ++l) {
        _Float16* wq = WT + (size_t)l * 7077888;
        _Float16* wk = wq + HH;
        _Float16* wv = wk + HH;
        _Float16* wp = wv + HH;
        _Float16* w1 = wp + HH;
        _Float16* w2 = w1 + HF;

        ln_kernel<<<kM / 8, 256, 0, stream>>>(X, ln1_g + l * kH, ln1_b + l * kH, HN);
        gemm_wmma<<<gH, 256, 0, stream>>>(HN, wq, bq + l * kH, Qb, nullptr, kH, kH, 3);
        gemm_wmma<<<gH, 256, 0, stream>>>(HN, wk, bk + l * kH, Kb, nullptr, kH, kH, 3);
        gemm_wmma<<<gH, 256, 0, stream>>>(HN, wv, bv + l * kH, VT, nullptr, kH, kH, 4);
        attn_kernel<<<kB * kNH * (kS / 16), 32, 0, stream>>>(Qb, Kb, VT, MADD, Y);
        gemm_wmma<<<gH, 256, 0, stream>>>(Y, wp, bp + l * kH, HN, X, kH, kH, 2);
        ln_kernel<<<kM / 8, 256, 0, stream>>>(X, ln2_g + l * kH, ln2_b + l * kH, HN);
        gemm_wmma<<<gF, 256, 0, stream>>>(HN, w1, b1 + l * kFF, A1, nullptr, kFF, kH, 1);
        gemm_wmma<<<gH, 256, 0, stream>>>(A1, w2, b2 + l * kH, HN, X, kH, kFF, 2);
    }

    predict_kernel<<<kB * kT, 32, 0, stream>>>(X, mmask, W_pred, b_pred, (float*)d_out);
}
